// Attention_37744172597821
// MI455X (gfx1250) — matrix operand order repaired and checked by reading, not compiled, not run on hardware
//
#include <hip/hip_runtime.h>
#include <hip/hip_bf16.h>

typedef __attribute__((ext_vector_type(16))) __bf16 v16bf;
typedef __attribute__((ext_vector_type(8)))  __bf16 v8bf;
typedef __attribute__((ext_vector_type(8)))  float  v8f;
typedef __attribute__((ext_vector_type(4)))  float  v4f;
typedef __attribute__((ext_vector_type(4)))  unsigned int u32x4;
typedef __attribute__((ext_vector_type(8)))  int    i32x8;
typedef __attribute__((ext_vector_type(4)))  int    i32x4;

#define B_   4
#define N_   1024
#define DIM_ 1024
#define H_   16
#define D_   64
#define M_   4
#define J_   (N_ + M_)   // 1028

__device__ __forceinline__ __bf16 f2bf(float f) {
    union { float f; unsigned u; } x; x.f = f;
    unsigned r = x.u + 0x7FFFu + ((x.u >> 16) & 1u);   // RNE
    unsigned short hs = (unsigned short)(r >> 16);
    __bf16 b;
    __builtin_memcpy(&b, &hs, sizeof(b));
    return b;
}

__device__ __forceinline__ v8bf pack8(v4f a0, v4f a1) {
    v8bf p;
    #pragma unroll
    for (int t = 0; t < 4; ++t) { p[t] = f2bf(a0[t]); p[t + 4] = f2bf(a1[t]); }
    return p;
}

// 16 contiguous bf16 at p -> one fragment register set (2 x ds/global b128)
__device__ __forceinline__ v16bf ld_frag16(const __bf16* p) {
    v8bf lo = *(const v8bf*)p;
    v8bf hh = *(const v8bf*)(p + 8);
    return __builtin_shufflevector(lo, hh, 0,1,2,3,4,5,6,7,8,9,10,11,12,13,14,15);
}
// A fragment from a row-major row (lane layout: e0-15 at k=hi*16+e)
__device__ __forceinline__ v16bf ld_frag_a(const __bf16* row, int hi) {
    v8bf lo = *(const v8bf*)(row + hi * 16);
    v8bf hh = *(const v8bf*)(row + hi * 16 + 8);
    return __builtin_shufflevector(lo, hh, 0,1,2,3,4,5,6,7,8,9,10,11,12,13,14,15);
}

// ---------------------------------------------------------------------------
// Kernel 0: tiled transpose + f32->bf16 convert:  in [R][C] f32 -> out [C][R] bf16
// ---------------------------------------------------------------------------
__global__ __launch_bounds__(256) void transpose_bf16_kernel(
    const float* __restrict__ in, __bf16* __restrict__ out,
    int R, int C, unsigned long long in_stride, unsigned long long out_stride)
{
    __shared__ float tile[32][33];
    const int tid = threadIdx.x;
    const float* src = in + (size_t)blockIdx.z * in_stride;
    __bf16* dst = out + (size_t)blockIdx.z * out_stride;
    const int r0 = blockIdx.x * 32, c0 = blockIdx.y * 32;

    for (int g = tid; g < 32 * 8; g += 256) {
        int rr = g >> 3, cc = (g & 7) * 4;
        v4f v = *(const v4f*)&src[(size_t)(r0 + rr) * C + c0 + cc];
        tile[rr][cc] = v[0]; tile[rr][cc + 1] = v[1];
        tile[rr][cc + 2] = v[2]; tile[rr][cc + 3] = v[3];
    }
    __syncthreads();
    for (int g = tid; g < 32 * 4; g += 256) {
        int cc = g >> 2, rb = (g & 3) * 8;
        v8bf p;
        #pragma unroll
        for (int t = 0; t < 8; ++t) p[t] = f2bf(tile[rb + t][cc]);
        *(v8bf*)&dst[(size_t)(c0 + cc) * R + r0 + rb] = p;
    }
}

// ---------------------------------------------------------------------------
// Kernel 1: QKV projection  Y = X @ W, W given pre-transposed bf16 [n][k]
// ---------------------------------------------------------------------------
__global__ __launch_bounds__(128) void qkv_gemm_kernel(
    const float* __restrict__ X,
    const __bf16* __restrict__ WqT, const __bf16* __restrict__ WkT, const __bf16* __restrict__ WvT,
    float* __restrict__ q_ws, float* __restrict__ k_ws, float* __restrict__ v_ws)
{
    __shared__ __bf16 Xs[64][32];    // [m][k]
    __shared__ __bf16 WsT[64][32];   // [n][k]

    const int tid = threadIdx.x, wv = tid >> 5, lane = tid & 31;
    const int ln = lane & 15, hi = lane >> 4;
    const int m0 = blockIdx.x * 64;
    const int n0 = blockIdx.y * 64;
    const int z  = blockIdx.z;

    const __bf16* Wt = (z == 0) ? WqT : (z == 1) ? WkT : WvT;
    float*       dst = (z == 0) ? q_ws : (z == 1) ? k_ws : v_ws;

    v8f acc[4];
    #pragma unroll
    for (int t = 0; t < 4; ++t)
        #pragma unroll
        for (int e = 0; e < 8; ++e) acc[t][e] = 0.f;

    for (int kb = 0; kb < DIM_; kb += 32) {
        __syncthreads();
        // stage X tile (f32 -> bf16, vectorized b128 in / b128 out)
        for (int g = tid; g < 64 * 4; g += 128) {
            int r = g >> 2, c0 = (g & 3) * 8;
            const float* s = &X[(size_t)(m0 + r) * DIM_ + kb + c0];
            *(v8bf*)&Xs[r][c0] = pack8(*(const v4f*)s, *(const v4f*)(s + 4));
        }
#if __has_builtin(__builtin_amdgcn_tensor_load_to_lds)
        // Tensor Data Mover: DMA the 64x32 bf16 weight tile straight into LDS.
        if (wv == 0) {
            unsigned long long ga =
                (unsigned long long)(uintptr_t)(Wt + (size_t)n0 * DIM_ + kb);
            unsigned lds = (unsigned)(uintptr_t)(&WsT[0][0]);
            u32x4 g0;
            g0[0] = 1u;                                   // count=1, user mode
            g0[1] = lds;                                  // lds_addr
            g0[2] = (unsigned)ga;                         // global_addr[31:0]
            g0[3] = (unsigned)((ga >> 32) & 0x01FFFFFFull) | (2u << 30); // addr hi | type=2
            i32x8 g1;
            g1[0] = (1 << 16);                            // data_size = 2 bytes
            g1[1] = (int)((DIM_ & 0xFFFF) << 16);         // tensor_dim0 lo
            g1[2] = (int)(((unsigned)DIM_ >> 16) | ((DIM_ & 0xFFFF) << 16)); // dim0 hi | dim1 lo
            g1[3] = (int)(((unsigned)DIM_ >> 16) | (32u << 16)); // dim1 hi | tile_dim0=32
            g1[4] = 64;                                   // tile_dim1=64, tile_dim2=0
            g1[5] = DIM_;                                 // tensor_dim0_stride = 1024
            g1[6] = 0; g1[7] = 0;
            i32x4 gz;
            gz[0] = 0; gz[1] = 0; gz[2] = 0; gz[3] = 0;
            i32x8 gz8;
            #pragma unroll
            for (int t = 0; t < 8; ++t) gz8[t] = 0;
            __builtin_amdgcn_tensor_load_to_lds(g0, g1, gz, gz, gz8, 0);
            __builtin_amdgcn_s_wait_tensorcnt(0);
        }
#else
        for (int g = tid; g < 64 * 2; g += 128) {
            int r = g >> 1, c0 = (g & 1) * 16;
            const v8bf* s = (const v8bf*)&Wt[(size_t)(n0 + r) * DIM_ + kb + c0];
            *(v8bf*)&WsT[r][c0]     = s[0];
            *(v8bf*)&WsT[r][c0 + 8] = s[1];
        }
#endif
        if (kb + 32 < DIM_)
            __builtin_prefetch(&X[(size_t)(m0 + (tid & 63)) * DIM_ + kb + 32], 0, 3);
        __syncthreads();

        v16bf a = ld_frag_a(&Xs[wv * 16 + ln][0], hi);
        #pragma unroll
        for (int t = 0; t < 4; ++t) {
            v16bf b = ld_frag16(&WsT[t * 16 + ln][hi * 16]);
            acc[t] = __builtin_amdgcn_wmma_f32_16x16x32_bf16(
                false, a, false, b, (short)0, acc[t], false, false);
        }
    }

    #pragma unroll
    for (int t = 0; t < 4; ++t)
        #pragma unroll
        for (int r = 0; r < 8; ++r) {
            int grow = m0 + wv * 16 + r + hi * 8;     // b*1024 + i
            int c    = n0 + t * 16 + ln;              // h*64 + dd
            int bb = grow >> 10, ii = grow & 1023;
            int hh = c >> 6,     dd = c & 63;
            dst[(((size_t)bb * H_ + hh) * N_ + ii) * D_ + dd] = acc[t][r];
        }
}

// ---------------------------------------------------------------------------
// Kernel 2: dots = scale * Q K^T  (j >= m part) -> pre region
// ---------------------------------------------------------------------------
__global__ __launch_bounds__(128) void dots_kernel(
    const float* __restrict__ q_ws, const float* __restrict__ k_ws,
    float* __restrict__ pre)
{
    __shared__ __bf16 Qs[64][64];
    __shared__ __bf16 Ks[64][64];

    const int tid = threadIdx.x, wv = tid >> 5, lane = tid & 31;
    const int ln = lane & 15, hi = lane >> 4;
    const int i0 = blockIdx.x * 64;
    const int j0 = blockIdx.y * 64;
    const int bh = blockIdx.z;

    const float* qb = q_ws + (size_t)bh * N_ * D_;
    const float* kb = k_ws + (size_t)bh * N_ * D_;

    for (int g = tid; g < 64 * 8; g += 128) {
        int r = g >> 3, c0 = (g & 7) * 8;
        const float* sq = &qb[(size_t)(i0 + r) * D_ + c0];
        const float* sk = &kb[(size_t)(j0 + r) * D_ + c0];
        *(v8bf*)&Qs[r][c0] = pack8(*(const v4f*)sq, *(const v4f*)(sq + 4));
        *(v8bf*)&Ks[r][c0] = pack8(*(const v4f*)sk, *(const v4f*)(sk + 4));
    }
    __syncthreads();

    v8f acc[4];
    #pragma unroll
    for (int t = 0; t < 4; ++t)
        #pragma unroll
        for (int e = 0; e < 8; ++e) acc[t][e] = 0.f;

    #pragma unroll
    for (int dc = 0; dc < 64; dc += 32) {
        v16bf a = ld_frag_a(&Qs[wv * 16 + ln][dc], hi);
        #pragma unroll
        for (int t = 0; t < 4; ++t) {
            v16bf b = ld_frag16(&Ks[t * 16 + ln][dc + hi * 16]);  // K rows are contraction-contiguous
            acc[t] = __builtin_amdgcn_wmma_f32_16x16x32_bf16(
                false, a, false, b, (short)0, acc[t], false, false);
        }
    }

    #pragma unroll
    for (int t = 0; t < 4; ++t)
        #pragma unroll
        for (int r = 0; r < 8; ++r) {
            int ii = i0 + wv * 16 + r + hi * 8;
            int jj = j0 + t * 16 + ln;
            pre[((size_t)bh * N_ + ii) * J_ + M_ + jj] = acc[t][r] * 0.125f;
        }
}

// memory-key columns j = 0..m-1
__global__ __launch_bounds__(256) void dots_mem_kernel(
    const float* __restrict__ q_ws, const float* __restrict__ mem_k,
    float* __restrict__ pre)
{
    int gid = blockIdx.x * 256 + threadIdx.x;
    if (gid >= B_ * H_ * N_) return;
    int bh = gid >> 10, i = gid & 1023, h = bh & 15;
    const float* qrow = q_ws + ((size_t)bh * N_ + i) * D_;
    #pragma unroll
    for (int j = 0; j < M_; ++j) {
        const float* kr = mem_k + (size_t)(h * M_ + j) * D_;
        float s = 0.f;
        for (int d = 0; d < D_; ++d) s += qrow[d] * kr[d];
        pre[((size_t)bh * N_ + i) * J_ + j] = s * 0.125f;
    }
}

// ---------------------------------------------------------------------------
// Kernel 3a: pre-softmax head mix + causal mask -> post region (raw logits)
// ---------------------------------------------------------------------------
__global__ __launch_bounds__(256) void mix_mask_kernel(
    const float* __restrict__ pre, const float* __restrict__ pre_proj,
    float* __restrict__ post)
{
    __shared__ float Ps[16][16];
    __shared__ float chunk[16][257];
    const int tid = threadIdx.x;
    const int bi = blockIdx.x;               // b*1024 + i
    const int b = bi >> 10, i = bi & 1023;

    Ps[tid >> 4][tid & 15] = pre_proj[tid];
    __syncthreads();

    for (int jb = 0; jb < J_; jb += 256) {
        for (int idx = tid; idx < 16 * 256; idx += 256) {
            int h = idx >> 8, jj = idx & 255;
            int j = jb + jj;
            if (j < J_)
                chunk[h][jj] = pre[(((size_t)(b * 16 + h)) * N_ + i) * J_ + j];
        }
        __syncthreads();
        int j = jb + tid;
        if (j < J_) {
            float vals[16];
            #pragma unroll
            for (int h = 0; h < 16; ++h) vals[h] = chunk[h][tid];
            bool masked = (j >= M_) && ((j - M_) > i);
            #pragma unroll
            for (int k = 0; k < 16; ++k) {
                float s = 0.f;
                #pragma unroll
                for (int h = 0; h < 16; ++h) s += vals[h] * Ps[h][k];
                post[(((size_t)(b * 16 + k)) * N_ + i) * J_ + j] = masked ? -__FLT_MAX__ : s;
            }
        }
        __syncthreads();
    }
}

// Kernel 3b: in-place row softmax over j
__global__ __launch_bounds__(128) void softmax_kernel(float* __restrict__ post)
{
    __shared__ float red[128];
    float* p = post + (size_t)blockIdx.x * J_;
    const int tid = threadIdx.x;

    float mx = -__FLT_MAX__;
    for (int j = tid; j < J_; j += 128) mx = fmaxf(mx, p[j]);
    red[tid] = mx; __syncthreads();
    for (int s = 64; s > 0; s >>= 1) {
        if (tid < s) red[tid] = fmaxf(red[tid], red[tid + s]);
        __syncthreads();
    }
    mx = red[0]; __syncthreads();

    float sum = 0.f;
    for (int j = tid; j < J_; j += 128) sum += __expf(p[j] - mx);
    red[tid] = sum; __syncthreads();
    for (int s = 64; s > 0; s >>= 1) {
        if (tid < s) red[tid] += red[tid + s];
        __syncthreads();
    }
    float inv = 1.f / red[0];
    for (int j = tid; j < J_; j += 128) p[j] = __expf(p[j] - mx) * inv;
}

// ---------------------------------------------------------------------------
// Kernel 4: fused post-softmax head mix + attn*V -> o1 [b, n, h*d]
// 16 waves/block; wave = output head; vT is bf16 [bh][d][n]
// ---------------------------------------------------------------------------
__global__ __launch_bounds__(512) void av_kernel(
    const float* __restrict__ post, const float* __restrict__ post_proj,
    const __bf16* __restrict__ vT, const float* __restrict__ mem_v,
    float* __restrict__ o1)
{
    __shared__ float Ps[16][16];
    __shared__ float stage[16][16][36];      // [h][i_local][jj] (padded)

    const int tid = threadIdx.x;
    const int k = tid >> 5;                  // wave = output head
    const int lane = tid & 31, ln = lane & 15, hi = lane >> 4;
    const int b  = blockIdx.x >> 6;
    const int i0 = (blockIdx.x & 63) * 16;

    if (tid < 256) Ps[tid >> 4][tid & 15] = post_proj[tid];
    __syncthreads();

    float pk[16];
    #pragma unroll
    for (int h = 0; h < 16; ++h) pk[h] = Ps[h][k];

    v8f acc[4];
    #pragma unroll
    for (int t = 0; t < 4; ++t)
        #pragma unroll
        for (int e = 0; e < 8; ++e) acc[t][e] = 0.f;

    // memory-value columns (j < m) via VALU
    for (int g = tid; g < 16 * 16; g += 512) {
        int h = g >> 4, il = g & 15;
        v4f v = *(const v4f*)&post[(((size_t)(b * 16 + h)) * N_ + i0 + il) * J_];
        *(v4f*)&stage[h][il][0] = v;
    }
    __syncthreads();
    #pragma unroll
    for (int r = 0; r < 8; ++r) {
        int il = r + hi * 8;
        #pragma unroll
        for (int j = 0; j < M_; ++j) {
            float a = 0.f;
            #pragma unroll
            for (int h = 0; h < 16; ++h) a += stage[h][il][j] * pk[h];
            #pragma unroll
            for (int t = 0; t < 4; ++t)
                acc[t][r] += a * mem_v[(size_t)(k * M_ + j) * D_ + t * 16 + ln];
        }
    }
    __syncthreads();

    const __bf16* vtb = vT + (size_t)(b * 16 + k) * D_ * N_;

    for (int jc = 0; jc < 32; ++jc) {
        // stage post[b, h, i0+il, 4 + jc*32 + jj]  (v4f b128 loads/stores)
        for (int g = tid; g < 16 * 16 * 8; g += 512) {
            int h = g >> 7, il = (g >> 3) & 15, j4 = (g & 7) * 4;
            v4f v = *(const v4f*)
                &post[(((size_t)(b * 16 + h)) * N_ + i0 + il) * J_ + M_ + jc * 32 + j4];
            *(v4f*)&stage[h][il][j4] = v;
        }
        __syncthreads();

        // mixed-attn A fragment: vectorized LDS reads, 16 FMAs per element
        float sacc[16];
        #pragma unroll
        for (int e = 0; e < 16; ++e) sacc[e] = 0.f;
        #pragma unroll
        for (int h = 0; h < 16; ++h) {
            const float w = pk[h];
            const v4f x0 = *(const v4f*)&stage[h][ln][hi * 16];
            const v4f x1 = *(const v4f*)&stage[h][ln][hi * 16 + 4];
            const v4f x2 = *(const v4f*)&stage[h][ln][hi * 16 + 8];
            const v4f x3 = *(const v4f*)&stage[h][ln][hi * 16 + 12];
            #pragma unroll
            for (int t = 0; t < 4; ++t) {
                sacc[t]      += x0[t] * w;
                sacc[t + 4]  += x1[t] * w;
                sacc[t + 8]  += x2[t] * w;
                sacc[t + 12] += x3[t] * w;
            }
        }
        v16bf a;
        #pragma unroll
        for (int e = 0; e < 16; ++e) a[e] = f2bf(sacc[e]);

        #pragma unroll
        for (int t = 0; t < 4; ++t) {
            // V^T row d = t*16+ln, contiguous j range -> 2 x global b128
            v16bf bb = ld_frag16(&vtb[(size_t)(t * 16 + ln) * N_ + jc * 32 + hi * 16]);
            acc[t] = __builtin_amdgcn_wmma_f32_16x16x32_bf16(
                false, a, false, bb, (short)0, acc[t], false, false);
        }
        __syncthreads();
    }

    #pragma unroll
    for (int t = 0; t < 4; ++t)
        #pragma unroll
        for (int r = 0; r < 8; ++r) {
            int ii = i0 + r + hi * 8;
            o1[((size_t)(b * N_ + ii)) * (H_ * D_) + k * D_ + t * 16 + ln] = acc[t][r];
        }
}

// ---------------------------------------------------------------------------
// Kernel 5: out = o1 @ Wout + bout  (Wout pre-transposed bf16 [n][k])
// ---------------------------------------------------------------------------
__global__ __launch_bounds__(128) void out_proj_kernel(
    const float* __restrict__ A, const __bf16* __restrict__ WoutT,
    const float* __restrict__ bias, float* __restrict__ Y)
{
    __shared__ __bf16 Xs[64][32];
    __shared__ __bf16 WsT[64][32];

    const int tid = threadIdx.x, wv = tid >> 5, lane = tid & 31;
    const int ln = lane & 15, hi = lane >> 4;
    const int m0 = blockIdx.x * 64;
    const int n0 = blockIdx.y * 64;

    v8f acc[4];
    #pragma unroll
    for (int t = 0; t < 4; ++t)
        #pragma unroll
        for (int e = 0; e < 8; ++e) acc[t][e] = 0.f;

    for (int kb = 0; kb < DIM_; kb += 32) {
        __syncthreads();
        for (int g = tid; g < 64 * 4; g += 128) {
            int r = g >> 2, c0 = (g & 3) * 8;
            const float* s = &A[(size_t)(m0 + r) * DIM_ + kb + c0];
            *(v8bf*)&Xs[r][c0] = pack8(*(const v4f*)s, *(const v4f*)(s + 4));
        }
        for (int g = tid; g < 64 * 2; g += 128) {
            int r = g >> 1, c0 = (g & 1) * 16;
            const v8bf* s = (const v8bf*)&WoutT[(size_t)(n0 + r) * DIM_ + kb + c0];
            *(v8bf*)&WsT[r][c0]     = s[0];
            *(v8bf*)&WsT[r][c0 + 8] = s[1];
        }
        __syncthreads();

        v16bf a = ld_frag_a(&Xs[wv * 16 + ln][0], hi);
        #pragma unroll
        for (int t = 0; t < 4; ++t) {
            v16bf b = ld_frag16(&WsT[t * 16 + ln][hi * 16]);
            acc[t] = __builtin_amdgcn_wmma_f32_16x16x32_bf16(
                false, a, false, b, (short)0, acc[t], false, false);
        }
    }

    #pragma unroll
    for (int t = 0; t < 4; ++t)
        #pragma unroll
        for (int r = 0; r < 8; ++r) {
            int grow = m0 + wv * 16 + r + hi * 8;
            int c    = n0 + t * 16 + ln;
            Y[(size_t)grow * DIM_ + c] = acc[t][r] + bias[c];
        }
}

// ---------------------------------------------------------------------------
extern "C" void kernel_launch(void* const* d_in, const int* in_sizes, int n_in,
                              void* d_out, int out_size, void* d_ws, size_t ws_size,
                              hipStream_t stream) {
    const float* x         = (const float*)d_in[0];
    const float* Wq        = (const float*)d_in[1];
    const float* Wk        = (const float*)d_in[2];
    const float* Wv        = (const float*)d_in[3];
    const float* mem_k     = (const float*)d_in[4];
    const float* mem_v     = (const float*)d_in[5];
    const float* pre_proj  = (const float*)d_in[6];
    const float* post_proj = (const float*)d_in[7];
    const float* Wout      = (const float*)d_in[8];
    const float* bout      = (const float*)d_in[9];

    float* out  = (float*)d_out;                                   // [4,1024,1024]
    float* pre  = out + (size_t)B_ * N_ * DIM_;                    // [4,16,1024,1028]
    float* post = pre + (size_t)B_ * H_ * N_ * J_;                 // [4,16,1024,1028]

    char* ws = (char*)d_ws;
    const size_t QKV = (size_t)B_ * H_ * N_ * D_;                  // 4,194,304 elems
    float*  q_ws  = (float*)(ws);
    float*  k_ws  = q_ws + QKV;
    float*  v_ws  = k_ws + QKV;
    float*  o1    = v_ws + QKV;                                    // [4096,1024]
    __bf16* WqT   = (__bf16*)(o1 + (size_t)B_ * N_ * DIM_);        // each 1M bf16
    __bf16* WkT   = WqT + (size_t)DIM_ * DIM_;
    __bf16* WvT   = WkT + (size_t)DIM_ * DIM_;
    __bf16* WoutT = WvT + (size_t)DIM_ * DIM_;
    __bf16* vT    = WoutT + (size_t)DIM_ * DIM_;                   // [bh][64][1024]

    // weight transposes -> bf16 [n][k]
    transpose_bf16_kernel<<<dim3(32, 32, 1), 256, 0, stream>>>(Wq,  WqT,  DIM_, DIM_, 0, 0);
    transpose_bf16_kernel<<<dim3(32, 32, 1), 256, 0, stream>>>(Wk,  WkT,  DIM_, DIM_, 0, 0);
    transpose_bf16_kernel<<<dim3(32, 32, 1), 256, 0, stream>>>(Wv,  WvT,  DIM_, DIM_, 0, 0);
    transpose_bf16_kernel<<<dim3(32, 32, 1), 256, 0, stream>>>(Wout, WoutT, DIM_, DIM_, 0, 0);

    qkv_gemm_kernel<<<dim3(64, 16, 3), 128, 0, stream>>>(x, WqT, WkT, WvT, q_ws, k_ws, v_ws);

    // V transpose per (b,h): [1024 j][64 d] -> bf16 [64 d][1024 j]
    transpose_bf16_kernel<<<dim3(32, 2, 64), 256, 0, stream>>>(
        v_ws, vT, N_, D_, (unsigned long long)(N_ * D_), (unsigned long long)(N_ * D_));

    dots_kernel<<<dim3(16, 16, 64), 128, 0, stream>>>(q_ws, k_ws, pre);
    dots_mem_kernel<<<dim3((B_ * H_ * N_ + 255) / 256), 256, 0, stream>>>(q_ws, mem_k, pre);
    mix_mask_kernel<<<dim3(B_ * N_), 256, 0, stream>>>(pre, pre_proj, post);
    softmax_kernel<<<dim3(B_ * H_ * N_), 128, 0, stream>>>(post);
    av_kernel<<<dim3(B_ * 64), 512, 0, stream>>>(post, post_proj, vT, mem_v, o1);
    out_proj_kernel<<<dim3(64, 16), 128, 0, stream>>>(o1, WoutT, bout, out);
}
